// SelfAttentionBlockWithRoPE_66898410602556
// MI455X (gfx1250) — matrix operand order repaired and checked by reading, not compiled, not run on hardware
//
#include <hip/hip_runtime.h>
#include <hip/hip_bf16.h>
#include <math.h>

// ---- CDNA5 WMMA types ----
typedef __attribute__((ext_vector_type(16))) __bf16 v16bf;
typedef __attribute__((ext_vector_type(8)))  float  v8f;

#define BDIM 1024
#define NHEADS 16
#define HDIM 64
#define SEQ 2048
#define NB 2
#define NTOK (NB * SEQ)      // 4096 rows
#define HID 4096

// packed f32x2 -> bf16x2 via the hardware packed converter (1 VALU op)
__device__ __forceinline__ unsigned int cvt_pk_bf16(float a, float b) {
    unsigned int r;
    asm("v_cvt_pk_bf16_f32 %0, %1, %2" : "=v"(r) : "v"(a), "v"(b));
    return r;
}
// fp32 -> bf16 bits (scalar path)
__device__ __forceinline__ unsigned short f2bits(float f) {
    union { __hip_bfloat16 h; unsigned short s; } c;
    c.h = __float2bfloat16(f);
    return c.s;
}
// pack 4 consecutive bf16 (2 packed cvts) and store as one 8-byte LDS write
__device__ __forceinline__ void pack4(unsigned short* dst, float4 f) {
    union { unsigned int w[2]; unsigned long long u; } p;
    p.w[0] = cvt_pk_bf16(f.x, f.y);
    p.w[1] = cvt_pk_bf16(f.z, f.w);
    *(unsigned long long*)dst = p.u;
}

// ---- pure-VALU 16-lane butterfly reductions via DPP ----
template<int CTRL>
__device__ __forceinline__ float dpp_step(float x) {
    int s = __builtin_amdgcn_update_dpp(0, __builtin_bit_cast(int, x), CTRL, 0xF, 0xF, true);
    return __builtin_bit_cast(float, s);
}
__device__ __forceinline__ float red_max16(float x) {
    x = fmaxf(x, dpp_step<0xB1>(x));    // quad_perm [1,0,3,2]  (xor 1)
    x = fmaxf(x, dpp_step<0x4E>(x));    // quad_perm [2,3,0,1]  (xor 2)
    x = fmaxf(x, dpp_step<0x141>(x));   // row_half_mirror      (combine quads in 8)
    x = fmaxf(x, dpp_step<0x140>(x));   // row_mirror           (combine 8s in 16)
    return x;
}
__device__ __forceinline__ float red_sum16(float x) {
    x += dpp_step<0xB1>(x);
    x += dpp_step<0x4E>(x);
    x += dpp_step<0x141>(x);
    x += dpp_step<0x140>(x);
    return x;
}

// ---------------------------------------------------------------------------
// LayerNorm: one 256-thread block per row of 1024, float4 traffic
// ---------------------------------------------------------------------------
__global__ __launch_bounds__(256)
void ln_kernel(const float* __restrict__ x, const float* __restrict__ w,
               const float* __restrict__ b, float* __restrict__ out) {
    const int row = blockIdx.x;
    const float4 v = ((const float4*)(x + (size_t)row * BDIM))[threadIdx.x];
    float s  = v.x + v.y + v.z + v.w;
    float s2 = v.x * v.x + v.y * v.y + v.z * v.z + v.w * v.w;
#pragma unroll
    for (int off = 1; off < 32; off <<= 1) {
        s  += __shfl_xor(s,  off, 32);
        s2 += __shfl_xor(s2, off, 32);
    }
    __shared__ float ss[8], ss2[8];
    if ((threadIdx.x & 31) == 0) { ss[threadIdx.x >> 5] = s; ss2[threadIdx.x >> 5] = s2; }
    __syncthreads();
    s = 0.f; s2 = 0.f;
#pragma unroll
    for (int i = 0; i < 8; ++i) { s += ss[i]; s2 += ss2[i]; }
    const float mean = s * (1.0f / BDIM);
    const float var  = s2 * (1.0f / BDIM) - mean * mean;
    const float inv  = rsqrtf(var + 1e-5f);
    const float4 wv = ((const float4*)w)[threadIdx.x];
    const float4 bv = ((const float4*)b)[threadIdx.x];
    float4 o;
    o.x = (v.x - mean) * inv * wv.x + bv.x;
    o.y = (v.y - mean) * inv * wv.y + bv.y;
    o.z = (v.z - mean) * inv * wv.z + bv.z;
    o.w = (v.w - mean) * inv * wv.w + bv.w;
    ((float4*)(out + (size_t)row * BDIM))[threadIdx.x] = o;
}

// ---------------------------------------------------------------------------
// RoPE, in-place on q and k slices of qkv [NTOK, 3072]
// ---------------------------------------------------------------------------
__global__ void rope_kernel(float* __restrict__ qkv, const int* __restrict__ pos) {
    int idx = blockIdx.x * blockDim.x + threadIdx.x;
    if (idx >= NTOK * 2 * NHEADS * 32) return;
    const int d  = idx & 31;  idx >>= 5;
    const int h  = idx & 15;  idx >>= 4;
    const int qk = idx & 1;   idx >>= 1;
    const int tok = idx;
    const float p = (float)pos[tok];
    const float invf = __powf(10000.f, -(float)d * (1.0f / 32.0f));
    float sn, cs;
    __sincosf(p * invf, &sn, &cs);
    const size_t base = (size_t)tok * (3 * BDIM) + qk * BDIM + h * HDIM;
    const float t1 = qkv[base + d];
    const float t2 = qkv[base + d + 32];
    qkv[base + d]      = t1 * cs - t2 * sn;
    qkv[base + d + 32] = t2 * cs + t1 * sn;
}

// ---------------------------------------------------------------------------
// WMMA GEMM, block tile 64(M) x 128(N), 256 threads = 8 waves.
// LDS tiles live in WMMA fragment layout [lane][e] -> fragment loads are
// single contiguous 32B LDS reads (ds_load_b128 pairs).
// A staged row-wise (k-contiguous), W staged column-wise (4 k-rows/thread) so
// every LDS write is a packed ds_store_b64.
// MODE: 1 = +bias ; 2 = +bias,GELU ; 3 = +bias,+residual
// ---------------------------------------------------------------------------
template<int N, int K, int MODE>
__global__ __launch_bounds__(256)
void gemm_wmma(const float* __restrict__ A, const float* __restrict__ W,
               const float* __restrict__ bias, const float* resid,
               float* __restrict__ out) {
    __shared__ __align__(32) unsigned short AsF[4][32][16];  // [mSub][lane][e]
    __shared__ __align__(32) unsigned short BsF[8][32][16];  // [nSub(wave)][lane][e]
    const int m0   = blockIdx.x * 64;
    const int n0   = blockIdx.y * 128;
    const int tid  = threadIdx.x;
    const int wave = tid >> 5;
    const int lane = tid & 31;
    const int half = lane >> 4;
    const int l16  = lane & 15;

    v8f z = {};
    v8f acc[4];
#pragma unroll
    for (int ms = 0; ms < 4; ++ms) acc[ms] = z;

    for (int k0 = 0; k0 < K; k0 += 32) {
        // ---- stage A tile (64 x 32): row-wise float4, packed b64 LDS stores
#pragma unroll
        for (int it = 0; it < 2; ++it) {
            const int i  = tid + it * 256;        // float4 index 0..511
            const int r  = i >> 3;                // tile row 0..63
            const int c4 = (i & 7) * 4;           // k offset 0,4,..,28
            float4 f = *(const float4*)(A + (size_t)(m0 + r) * K + k0 + c4);
            const int hf = (c4 >> 3) & 1;
            const int e0 = (c4 & 7) + 8 * (c4 >> 4);
            pack4(&AsF[r >> 4][(r & 15) + 16 * hf][e0], f);
        }
        // ---- stage W tile (32 x 128): column-wise (4 k rows), coalesced
#pragma unroll
        for (int it = 0; it < 4; ++it) {
            const int g  = tid + it * 256;        // column-group 0..1023
            const int r4 = (g >> 7) * 4;          // k row base 0,4,...,28
            const int c  = g & 127;               // n offset
            const float* wp = W + (size_t)(k0 + r4) * N + n0 + c;
            float4 f;
            f.x = wp[0];
            f.y = wp[(size_t)N];
            f.z = wp[2 * (size_t)N];
            f.w = wp[3 * (size_t)N];
            pack4(&BsF[c >> 4][(c & 15) + 16 * ((r4 >> 3) & 1)][(r4 & 7) + 8 * (r4 >> 4)], f);
        }
        if (k0 + 32 < K)    // lookahead -> global_prefetch_b8
            __builtin_prefetch(W + (size_t)(k0 + 32 + (tid >> 3)) * N + n0 + (tid & 7) * 16, 0, 1);
        __syncthreads();

        const v16bf bf = *(const v16bf*)&BsF[wave][lane][0];
#pragma unroll
        for (int ms = 0; ms < 4; ++ms) {
            const v16bf af = *(const v16bf*)&AsF[ms][lane][0];
            acc[ms] = __builtin_amdgcn_wmma_f32_16x16x32_bf16(
                          false, af, false, bf, (short)0, acc[ms], false, false);
        }
        __syncthreads();
    }

    // epilogue: row = m0 + ms*16 + half*8 + r ; col = n0 + wave*16 + l16
    const int col = n0 + wave * 16 + l16;
    const float bc = bias[col];
#pragma unroll
    for (int ms = 0; ms < 4; ++ms)
#pragma unroll
        for (int r = 0; r < 8; ++r) {
            const int row = m0 + ms * 16 + half * 8 + r;
            float v = acc[ms][r] + bc;
            if (MODE == 2) v = 0.5f * v * (1.0f + erff(v * 0.70710678118654752f));
            if (MODE == 3) v += resid[(size_t)row * N + col];
            out[(size_t)row * N + col] = v;
        }
}

// ---------------------------------------------------------------------------
// Flash attention: 256 threads = 8 waves, each wave owns a 16-row Q tile of
// the same (b,h); the block cooperatively stages each 32x64 K and V chunk
// into LDS in fragment layout (coalesced loads, amortized 8x).
// ---------------------------------------------------------------------------
__global__ __launch_bounds__(256)
void attn_kernel(const float* __restrict__ qkv, float* __restrict__ attn_out) {
    const int qt0  = blockIdx.x;      // 128-row Q slab
    const int h    = blockIdx.y;
    const int b    = blockIdx.z;
    const int tid  = threadIdx.x;
    const int wave = tid >> 5;
    const int lane = tid & 31;
    const int half = lane >> 4;
    const int l16  = lane & 15;
    const int rowBase = qt0 * 128 + wave * 16;
    const size_t rs = 3 * BDIM;
    const float scale = 0.125f;

    __shared__ __align__(32) unsigned short KsF[2][2][32][16]; // [nt][ks][lane][e]
    __shared__ __align__(32) unsigned short VsF[4][32][16];    // [dt][lane][e]
    __shared__ __align__(32) unsigned short PsF[8][32][16];    // [wave][lane][e]

    const float* Qb = qkv + (size_t)(b * SEQ + rowBase) * rs + h * HDIM;
    const float* Kb = qkv + (size_t)(b * SEQ) * rs + BDIM + h * HDIM;
    const float* Vb = qkv + (size_t)(b * SEQ) * rs + 2 * BDIM + h * HDIM;

    // ---- Q fragments held in registers, vector loads + packed cvt
    v16bf qf[2];
#pragma unroll
    for (int ks = 0; ks < 2; ++ks) {
        const float4* q0 = (const float4*)(Qb + (size_t)l16 * rs + ks * 32 + half * 8);
        const float4* q1 = (const float4*)(Qb + (size_t)l16 * rs + ks * 32 + 16 + half * 8);
        const float4 a = q0[0], bb = q0[1], c = q1[0], d = q1[1];
        union { v16bf v; unsigned short s[16]; } q;
        pack4(&q.s[0],  a);
        pack4(&q.s[4],  bb);
        pack4(&q.s[8],  c);
        pack4(&q.s[12], d);
        qf[ks] = q.v;
    }

    float m[8], lsum[8];
    v8f z = {};
    v8f acc[4];
#pragma unroll
    for (int r = 0; r < 8; ++r) { m[r] = -1e30f; lsum[r] = 0.f; }
#pragma unroll
    for (int dt = 0; dt < 4; ++dt) acc[dt] = z;

    for (int t0 = 0; t0 < SEQ; t0 += 32) {
        // ---- stage K chunk (32 kv rows x 64 d): row-wise float4 (d-contig)
#pragma unroll
        for (int it = 0; it < 2; ++it) {
            const int i  = tid + it * 256;         // float4 idx 0..511
            const int t  = i >> 4;                 // kv row 0..31
            const int c4 = (i & 15) * 4;           // d 0..60
            float4 f = *(const float4*)(Kb + (size_t)(t0 + t) * rs + c4);
            const int ks = c4 >> 5;
            const int hf = (c4 >> 3) & 1;
            pack4(&KsF[t >> 4][ks][(t & 15) + 16 * hf][(c4 & 7) + 8 * ((c4 >> 4) & 1)], f);
        }
        // ---- stage V chunk (32 kv rows x 64 d): column-wise (4 kv rows)
#pragma unroll
        for (int it = 0; it < 2; ++it) {
            const int g  = tid + it * 256;         // 0..511
            const int t4 = (g >> 6) * 4;           // kv row base 0,4,..,28
            const int c  = g & 63;                 // d
            const float* vp = Vb + (size_t)(t0 + t4) * rs + c;
            float4 f;
            f.x = vp[0];
            f.y = vp[rs];
            f.z = vp[2 * rs];
            f.w = vp[3 * rs];
            pack4(&VsF[c >> 4][(c & 15) + 16 * ((t4 >> 3) & 1)][(t4 & 7) + 8 * (t4 >> 4)], f);
        }
        __syncthreads();

        // ---- scores: Q(16x64) @ K^T(64x32) -> two 16x16 tiles
        v8f s[2]; s[0] = z; s[1] = z;
#pragma unroll
        for (int nt = 0; nt < 2; ++nt)
#pragma unroll
            for (int ks = 0; ks < 2; ++ks) {
                const v16bf kf = *(const v16bf*)&KsF[nt][ks][lane][0];
                s[nt] = __builtin_amdgcn_wmma_f32_16x16x32_bf16(
                            false, qf[ks], false, kf, (short)0, s[nt], false, false);
            }

        // ---- online softmax per row (row = half*8 + r), DPP reductions
        float p0s[8], p1s[8];
#pragma unroll
        for (int r = 0; r < 8; ++r) {
            const float v0 = s[0][r] * scale;
            const float v1 = s[1][r] * scale;
            const float mx = red_max16(fmaxf(v0, v1));
            const float mnew  = fmaxf(m[r], mx);
            const float alpha = __expf(m[r] - mnew);
            const float p0 = __expf(v0 - mnew);
            const float p1 = __expf(v1 - mnew);
            const float psum = red_sum16(p0 + p1);
            lsum[r] = lsum[r] * alpha + psum;
            m[r] = mnew;
#pragma unroll
            for (int dt = 0; dt < 4; ++dt) acc[dt][r] *= alpha;
            p0s[r] = p0; p1s[r] = p1;
        }

        // ---- P: C-layout -> A-fragment layout via per-wave LDS buffer
        {
            const int hf2 = (l16 >> 3) & 1;
            const int eb  = l16 & 7;
#pragma unroll
            for (int r = 0; r < 8; ++r) {
                const int lp = half * 8 + r + 16 * hf2;
                PsF[wave][lp][eb]     = f2bits(p0s[r]);
                PsF[wave][lp][eb + 8] = f2bits(p1s[r]);
            }
        }
        // same-wave LDS ops are in order; no block barrier needed for PsF
        const v16bf pf = *(const v16bf*)&PsF[wave][lane][0];

        // ---- P(16x32) @ V(32x64)
#pragma unroll
        for (int dt = 0; dt < 4; ++dt) {
            const v16bf vf = *(const v16bf*)&VsF[dt][lane][0];
            acc[dt] = __builtin_amdgcn_wmma_f32_16x16x32_bf16(
                          false, pf, false, vf, (short)0, acc[dt], false, false);
        }
        __syncthreads();   // protect K/V tiles before next stage
    }

    // ---- write out: attn layout [b, s, h, d] = [NTOK, 1024]
#pragma unroll
    for (int dt = 0; dt < 4; ++dt)
#pragma unroll
        for (int r = 0; r < 8; ++r) {
            const int row = rowBase + half * 8 + r;
            const int col = h * HDIM + dt * 16 + l16;
            attn_out[(size_t)(b * SEQ + row) * BDIM + col] = acc[dt][r] / lsum[r];
        }
}

// ---------------------------------------------------------------------------
extern "C" void kernel_launch(void* const* d_in, const int* in_sizes, int n_in,
                              void* d_out, int out_size, void* d_ws, size_t ws_size,
                              hipStream_t stream) {
    const float* x      = (const float*)d_in[0];
    const int*   pos    = (const int*)d_in[1];
    const float* ln1_w  = (const float*)d_in[2];
    const float* ln1_b  = (const float*)d_in[3];
    const float* qkv_w  = (const float*)d_in[4];
    const float* qkv_b  = (const float*)d_in[5];
    const float* o_w    = (const float*)d_in[6];
    const float* o_b    = (const float*)d_in[7];
    const float* ln2_w  = (const float*)d_in[8];
    const float* ln2_b  = (const float*)d_in[9];
    const float* mlp_w1 = (const float*)d_in[10];
    const float* mlp_b1 = (const float*)d_in[11];
    const float* mlp_w2 = (const float*)d_in[12];
    const float* mlp_b2 = (const float*)d_in[13];
    float* out = (float*)d_out;

    // ws (floats): [xn | qkv | attn]; mlp hidden reuses qkv+attn (exact fit).
    float* ws      = (float*)d_ws;
    float* xn      = ws;                                  // 4096*1024
    float* qkvbuf  = ws + (size_t)NTOK * BDIM;            // 4096*3072
    float* attnbuf = qkvbuf + (size_t)NTOK * 3 * BDIM;    // 4096*1024
    float* hbuf    = qkvbuf;                              // 4096*4096 (reuse)

    ln_kernel<<<NTOK, 256, 0, stream>>>(x, ln1_w, ln1_b, xn);

    gemm_wmma<3 * BDIM, BDIM, 1><<<dim3(NTOK / 64, (3 * BDIM) / 128), 256, 0, stream>>>(
        xn, qkv_w, qkv_b, nullptr, qkvbuf);

    {
        const int nthr = NTOK * 2 * NHEADS * 32;
        rope_kernel<<<(nthr + 255) / 256, 256, 0, stream>>>(qkvbuf, pos);
    }

    attn_kernel<<<dim3(SEQ / 128, NHEADS, NB), 256, 0, stream>>>(qkvbuf, attnbuf);

    gemm_wmma<BDIM, BDIM, 3><<<dim3(NTOK / 64, BDIM / 128), 256, 0, stream>>>(
        attnbuf, o_w, o_b, x, out);

    ln_kernel<<<NTOK, 256, 0, stream>>>(out, ln2_w, ln2_b, xn);

    gemm_wmma<HID, BDIM, 2><<<dim3(NTOK / 64, HID / 128), 256, 0, stream>>>(
        xn, mlp_w1, mlp_b1, nullptr, hbuf);

    gemm_wmma<BDIM, HID, 3><<<dim3(NTOK / 64, BDIM / 128), 256, 0, stream>>>(
        hbuf, mlp_w2, mlp_b2, out, out);
}
